// Attention_53231824666850
// MI455X (gfx1250) — matrix operand order repaired and checked by reading, not compiled, not run on hardware
//
#include <hip/hip_runtime.h>

// Flash attention for B=4,H=16,S=2048,DK=64 (fp32 in/out), f16 WMMA compute
// with fp32 accumulation on gfx1250 (wave32, V_WMMA_F32_16X16X32_F16).
// Mask tile staged with CDNA5 async global->LDS copies (ASYNCcnt path).

#define B_DIM  4
#define H_DIM  16
#define S_DIM  2048
#define DK_DIM 64
#define BM     64          // query rows per workgroup
#define BN     64          // key/value rows per iteration
#define NW     4           // waves per workgroup (each wave: 16 query rows)
#define NKB    (S_DIM / BN)

typedef __attribute__((ext_vector_type(16))) _Float16 v16h;
typedef __attribute__((ext_vector_type(8)))  _Float16 v8h;
typedef __attribute__((ext_vector_type(8)))  float    v8f;

__device__ __forceinline__ float hmax16(float v) {
    v = fmaxf(v, __shfl_xor(v, 1, 16));
    v = fmaxf(v, __shfl_xor(v, 2, 16));
    v = fmaxf(v, __shfl_xor(v, 4, 16));
    v = fmaxf(v, __shfl_xor(v, 8, 16));
    return v;
}

__device__ __forceinline__ float hsum16(float v) {
    v += __shfl_xor(v, 1, 16);
    v += __shfl_xor(v, 2, 16);
    v += __shfl_xor(v, 4, 16);
    v += __shfl_xor(v, 8, 16);
    return v;
}

// Async 16B global->LDS copy, GVS addressing (mem = SGPR64 + VGPR_I32).
// VDST carries the per-lane LDS byte address (ISA 10.7 / 15.18.3 op 98).
__device__ __forceinline__ void async_load_b128(unsigned lds_addr,
                                                const void* base, int voff) {
    asm volatile("global_load_async_to_lds_b128 %0, %1, %2"
                 :
                 : "v"(lds_addr), "v"(voff), "s"(base)
                 : "memory");
}

__device__ __forceinline__ void wait_asynccnt0() {
    asm volatile("s_wait_asynccnt 0x0" ::: "memory");
}

// A-fragment (16x32 f16): lane m (= lane&15) holds row m; K elements
// {koff..koff+15} in halves 0-15 (contiguous, matching the B fragments),
// where koff = kbase + 16*(lane>=16).
__device__ __forceinline__ v16h load_a_frag(const _Float16* rowp, int koff) {
    v8h lo = *(const v8h*)(rowp + koff);
    v8h hi = *(const v8h*)(rowp + koff + 8);
    v16h a;
#pragma unroll
    for (int i = 0; i < 8; ++i) { a[i] = lo[i]; a[i + 8] = hi[i]; }
    return a;
}

__global__ __launch_bounds__(128)
void fa_wmma_kernel(const float* __restrict__ Q, const float* __restrict__ K,
                    const float* __restrict__ V, const unsigned char* __restrict__ M,
                    float* __restrict__ O)
{
    __shared__ __align__(16) _Float16 sQ[BM][DK_DIM];        // 8 KB
    __shared__ __align__(16) _Float16 sK[BN][DK_DIM];        // 8 KB
    __shared__ __align__(16) _Float16 sVt[DK_DIM][BN + 8];   // 9 KB (transposed V, padded)
    __shared__ __align__(16) unsigned char sM[BM][BN];       // 4 KB
    __shared__ __align__(16) _Float16 sP[NW][16][BN];        // 8 KB (per-wave P scratch)

    const int tid  = threadIdx.x;          // 0..127
    const int lane = tid & 31;
    const int wid  = tid >> 5;             // wave id 0..3
    const int qblk = blockIdx.x;
    const int bh   = blockIdx.y;
    const int b    = bh / H_DIM;

    const size_t head_off = (size_t)bh * S_DIM * DK_DIM;
    const int    qbase    = qblk * BM;

    const float* Qp = Q + head_off + (size_t)qbase * DK_DIM;
    float*       Op = O + head_off + (size_t)qbase * DK_DIM;
    const unsigned char* Mp = M + ((size_t)b * S_DIM + qbase) * S_DIM;

    // Per-thread mask-staging geometry (32 B per thread = 2 async b128)
    const int      mRow     = tid >> 1;
    const int      mOff     = (tid & 1) * 32;
    const unsigned mLds     = (unsigned)(size_t)&sM[mRow][mOff];
    const int      mGoffRow = mRow * S_DIM + mOff;

    // ---- stage Q into LDS as f16, scale folded in (1/sqrt(64) = 0.125, exact) ----
    {
        const float4* src = (const float4*)Qp;
        _Float16*     dst = &sQ[0][0];
#pragma unroll
        for (int i = 0; i < 8; ++i) {
            int e = i * 128 + tid;              // 1024 float4 total
            float4 v = src[e];
            dst[e * 4 + 0] = (_Float16)(v.x * 0.125f);
            dst[e * 4 + 1] = (_Float16)(v.y * 0.125f);
            dst[e * 4 + 2] = (_Float16)(v.z * 0.125f);
            dst[e * 4 + 3] = (_Float16)(v.w * 0.125f);
        }
    }
    __syncthreads();

    const int mrow   = lane & 15;           // row (A) / column (B) within 16-tile
    const int rowSel = (lane >> 4) << 3;    // C layout: upper half-wave owns rows +8
    const int aoff   = (lane >> 4) << 4;    // A-frag K offset for upper half-wave
    const int boff   = (lane >> 4) << 4;    // B-frag K offset for upper half-wave

    // Preload Q A-fragments (reused for all 32 K-blocks)
    const _Float16* qrow = &sQ[wid * 16 + mrow][0];
    v16h qa0 = load_a_frag(qrow, 0  + aoff);
    v16h qa1 = load_a_frag(qrow, 32 + aoff);

    v8f acc[4] = {};                        // output accumulator, 16 x 64 per wave
    float m_i[8], l_i[8];
#pragma unroll
    for (int r = 0; r < 8; ++r) { m_i[r] = -1e30f; l_i[r] = 0.0f; }

    for (int kb = 0; kb < NKB; ++kb) {
        __syncthreads();                    // protect LDS from previous iteration

        const float* Kp = K + head_off + (size_t)kb * BN * DK_DIM;
        const float* Vp = V + head_off + (size_t)kb * BN * DK_DIM;

        // ---- mask tile: async global->LDS byte copy (no conversion needed) ----
        {
            const unsigned char* Mb = Mp + (size_t)kb * BN;
            async_load_b128(mLds,      Mb, mGoffRow);
            async_load_b128(mLds + 16, Mb, mGoffRow + 16);
        }
        // ---- stage K tile (row major, f16) ----
        {
            const float4* src = (const float4*)Kp;
            _Float16*     dst = &sK[0][0];
#pragma unroll
            for (int i = 0; i < 8; ++i) {
                int e = i * 128 + tid;
                float4 v = src[e];
                dst[e * 4 + 0] = (_Float16)v.x;
                dst[e * 4 + 1] = (_Float16)v.y;
                dst[e * 4 + 2] = (_Float16)v.z;
                dst[e * 4 + 3] = (_Float16)v.w;
            }
        }
        // ---- stage V tile transposed: sVt[dk][s] ----
        {
            const float4* src = (const float4*)Vp;
#pragma unroll
            for (int i = 0; i < 8; ++i) {
                int f = i * 128 + tid;          // float4 index
                int s = f >> 4;                 // row (s) 0..63
                int d = (f & 15) << 2;          // dk 0..60 step 4
                float4 v = src[f];
                sVt[d + 0][s] = (_Float16)v.x;
                sVt[d + 1][s] = (_Float16)v.y;
                sVt[d + 2][s] = (_Float16)v.z;
                sVt[d + 3][s] = (_Float16)v.w;
            }
        }
        // ---- prefetch next K/V tiles into L2 (global_prefetch_b8) ----
        if (kb + 1 < NKB) {
            __builtin_prefetch(Kp + BN * DK_DIM + tid * 32, 0, 1);
            __builtin_prefetch(Vp + BN * DK_DIM + tid * 32, 0, 1);
        }
        wait_asynccnt0();                   // mask bytes landed in LDS
        __syncthreads();

        // ---- S = Q K^T  (4 column tiles of 16, K-dim 64 = 2 x 32) ----
        v8f c[4];
#pragma unroll
        for (int j = 0; j < 4; ++j) {
            const _Float16* brow = &sK[j * 16 + mrow][0];   // B column n = K row n
            v16h b0 = *(const v16h*)(brow + boff);
            v16h b1 = *(const v16h*)(brow + 32 + boff);
            v8f cj = {};
            cj = __builtin_amdgcn_wmma_f32_16x16x32_f16(false, qa0, false, b0,
                                                        (short)0, cj, false, false);
            cj = __builtin_amdgcn_wmma_f32_16x16x32_f16(false, qa1, false, b1,
                                                        (short)0, cj, false, false);
            c[j] = cj;
        }

        // ---- online softmax (row-wise over the 64 columns of this block) ----
#pragma unroll
        for (int r = 0; r < 8; ++r) {
            int qr = r + rowSel;                 // local row 0..15 within wave tile
            float s0 = sM[wid * 16 + qr][ 0 + mrow] ? -1e30f : c[0][r];
            float s1 = sM[wid * 16 + qr][16 + mrow] ? -1e30f : c[1][r];
            float s2 = sM[wid * 16 + qr][32 + mrow] ? -1e30f : c[2][r];
            float s3 = sM[wid * 16 + qr][48 + mrow] ? -1e30f : c[3][r];
            float mx = hmax16(fmaxf(fmaxf(s0, s1), fmaxf(s2, s3)));
            float mnew = fmaxf(m_i[r], mx);
            float sc   = __expf(m_i[r] - mnew);
            m_i[r] = mnew;
            float p0 = __expf(s0 - mnew);
            float p1 = __expf(s1 - mnew);
            float p2 = __expf(s2 - mnew);
            float p3 = __expf(s3 - mnew);
            sP[wid][qr][ 0 + mrow] = (_Float16)p0;
            sP[wid][qr][16 + mrow] = (_Float16)p1;
            sP[wid][qr][32 + mrow] = (_Float16)p2;
            sP[wid][qr][48 + mrow] = (_Float16)p3;
            float rs = hsum16(p0 + p1 + p2 + p3);
            l_i[r] = l_i[r] * sc + rs;
            acc[0][r] *= sc;
            acc[1][r] *= sc;
            acc[2][r] *= sc;
            acc[3][r] *= sc;
        }

        // ---- O += P V  (A = P from LDS, B = transposed V tile) ----
        const _Float16* prow = &sP[wid][mrow][0];
        v16h pa0 = load_a_frag(prow, 0  + aoff);
        v16h pa1 = load_a_frag(prow, 32 + aoff);
#pragma unroll
        for (int j = 0; j < 4; ++j) {
            const _Float16* vrow = &sVt[j * 16 + mrow][0];  // B column n = Vt row n
            v16h b0 = *(const v16h*)(vrow + boff);
            v16h b1 = *(const v16h*)(vrow + 32 + boff);
            acc[j] = __builtin_amdgcn_wmma_f32_16x16x32_f16(false, pa0, false, b0,
                                                            (short)0, acc[j], false, false);
            acc[j] = __builtin_amdgcn_wmma_f32_16x16x32_f16(false, pa1, false, b1,
                                                            (short)0, acc[j], false, false);
        }
    }

    // ---- epilogue: O /= l, store fp32 ----
#pragma unroll
    for (int j = 0; j < 4; ++j) {
#pragma unroll
        for (int r = 0; r < 8; ++r) {
            int qr = wid * 16 + r + rowSel;
            Op[(size_t)qr * DK_DIM + j * 16 + mrow] = acc[j][r] / l_i[r];
        }
    }
}

extern "C" void kernel_launch(void* const* d_in, const int* in_sizes, int n_in,
                              void* d_out, int out_size, void* d_ws, size_t ws_size,
                              hipStream_t stream) {
    const float*         Q  = (const float*)d_in[0];
    const float*         K  = (const float*)d_in[1];
    const float*         V  = (const float*)d_in[2];
    const unsigned char* M  = (const unsigned char*)d_in[3];  // jnp.bool_ = 1 byte
    float*               O  = (float*)d_out;

    dim3 grid(S_DIM / BM, B_DIM * H_DIM);   // 32 x 64 workgroups
    dim3 block(128);                         // 4 waves (wave32)
    fa_wmma_kernel<<<grid, block, 0, stream>>>(Q, K, V, M, O);
}
